// MultiHeadSelfAttention_17738214933039
// MI455X (gfx1250) — compile-verified, matrix operand order checked by reading, not run on hardware
//
#include <hip/hip_runtime.h>

// ---------------------------------------------------------------------------
// MultiHeadSelfAttention (B=16, S=4096, H=512, nh=1) for MI455X / gfx1250.
// GEMMs: v_wmma_f32_16x16x32_bf16.  Tile staging: Tensor Data Mover
// (tensor_load_to_lds) with hardware LDS row padding, double-buffered via
// TENSORcnt (s_wait_tensorcnt 2) so DMA overlaps WMMA.
// ---------------------------------------------------------------------------

typedef __attribute__((ext_vector_type(16))) __bf16 v16bf;
typedef __attribute__((ext_vector_type(4)))  __bf16 v4bf;
typedef __attribute__((ext_vector_type(8)))  float  v8f;
typedef __attribute__((ext_vector_type(4)))  unsigned int v4u;
typedef __attribute__((ext_vector_type(8)))  int    v8i;
typedef __attribute__((ext_vector_type(4)))  int    v4i;

#define HDIM   512
#define SDIM   4096
#define NB     16
#define MROWS  ((size_t)NB * SDIM)   // 65536 flattened rows of x
#define LDST   40                    // LDS row stride in halves (32 + 8 pad)

__device__ __forceinline__ v8f wmma_bf16(v16bf a, v16bf b, v8f c) {
  return __builtin_amdgcn_wmma_f32_16x16x32_bf16(false, a, false, b,
                                                 (short)0, c, false, false);
}

// ---------------------------------------------------------------------------
// TDM: DMA a rows x 32 bf16 tile (row-major, stride ld_elems) from global to
// LDS at byte offset lds_off.  Pad: 16 DWORDs (one 32-half row) then 4 DWORDs
// (8 halves) -> effective LDS stride = 40 halves = LDST.
// D# layout per CDNA5 ISA ch.8 (group0 128b, group1 256b).
// ---------------------------------------------------------------------------
__device__ __forceinline__ void tdm_load_2d(unsigned int lds_off,
                                            const void* gaddr,
                                            unsigned int rows,
                                            unsigned int ld_elems) {
  unsigned long long ga = (unsigned long long)gaddr;
  v4u g0;
  g0[0] = 1u;                                  // count=1 (valid user D#)
  g0[1] = lds_off;                             // lds_addr (bytes)
  g0[2] = (unsigned int)ga;                    // global_addr[31:0]
  g0[3] = (unsigned int)(ga >> 32) | (2u << 30); // global_addr[56:32] | type=2
  v8i g1;
  g1[0] = (int)((1u << 16)        // data_size = 1 -> 2 bytes (bf16)
              | (1u << 20)        // pad_enable
              | (3u << 22)        // pad_interval: code 3 = 16 DWORDs
              | (3u << 25));      // pad_amount:   code 3 = 4 DWORDs
  g1[1] = (int)(32u << 16);       // tensor_dim0 = 32 (bits [79:48], low half)
  g1[2] = (int)(rows << 16);      // tensor_dim0 hi = 0 | tensor_dim1[15:0]
  g1[3] = (int)(32u << 16);       // tensor_dim1 hi = 0 | tile_dim0 = 32
  g1[4] = (int)rows;              // tile_dim1 = rows | tile_dim2 = 0
  g1[5] = (int)ld_elems;          // tensor_dim0_stride[31:0]
  g1[6] = 0;                      // stride hi | tensor_dim1_stride lo
  g1[7] = 0;
  v4i z4 = {0, 0, 0, 0};
#if __has_include(<hip/amd_detail/amd_gfx1250_TDM.h>)
  v8i z8 = {0, 0, 0, 0, 0, 0, 0, 0};
  __builtin_amdgcn_tensor_load_to_lds(g0, g1, z4, z4, z8, 0);
#else
  __builtin_amdgcn_tensor_load_to_lds(g0, g1, z4, z4, 0);
#endif
}

__device__ __forceinline__ unsigned int lds_off_of(const void* p) {
  // Generic pointers to LDS carry the wave-relative LDS offset in [31:0].
  return (unsigned int)(unsigned long long)p;
}

// Load a 16x32 fragment (A layout; B uses the same pattern with lane = N col).
__device__ __forceinline__ v16bf load_frag(const __bf16* base, int lane) {
  int r  = lane & 15;
  int ko = (lane >> 4) << 3;        // 0 or 8
  const __bf16* p = base + r * LDST + ko;
  v4bf a0 = *(const v4bf*)(p);
  v4bf a1 = *(const v4bf*)(p + 4);
  v4bf a2 = *(const v4bf*)(p + 16);
  v4bf a3 = *(const v4bf*)(p + 20);
  v16bf f;
#pragma unroll
  for (int i = 0; i < 4; ++i) {
    f[i]      = a0[i];
    f[4 + i]  = a1[i];
    f[8 + i]  = a2[i];
    f[12 + i] = a3[i];
  }
  return f;
}

// Cooperative f32 -> bf16 tile load (for x), rows x 32, 256 threads.
__device__ __forceinline__ void tile_f32(__bf16* dst, const float* src,
                                         int ld, int rows, int tid) {
  int chunks = rows * 8;
  for (int c = tid; c < chunks; c += 256) {
    int r = c >> 3, col = (c & 7) << 2;
    float4 v = *(const float4*)(src + (size_t)r * ld + col);
    v4bf o;
    o.x = (__bf16)v.x; o.y = (__bf16)v.y; o.z = (__bf16)v.z; o.w = (__bf16)v.w;
    *(v4bf*)(dst + r * LDST + col) = o;
  }
}

// ---------------------------------------------------------------------------
// Kernel 1: transpose+convert the four weight matrices to bf16 [n][k] major.
// ---------------------------------------------------------------------------
__global__ __launch_bounds__(256) void prep_kernel(
    const float* Wq, const float* Wk, const float* Wv, const float* Wo,
    __bf16* WqT, __bf16* WkT, __bf16* WvT, __bf16* WoT) {
  int i = blockIdx.x * 256 + threadIdx.x;
  int h = i >> 9, n = i & 511;
  size_t s = (size_t)h * HDIM + n;
  size_t d = (size_t)n * HDIM + h;
  WqT[d] = (__bf16)Wq[s];
  WkT[d] = (__bf16)Wk[s];
  WvT[d] = (__bf16)Wv[s];
  WoT[d] = (__bf16)Wo[s];
}

// ---------------------------------------------------------------------------
// Kernel 2: fused QKV projection. x(f32) -> Qt,Kt (b,hd,S) + V (b,S,hd) bf16.
// Block tile 128(M) x 64(N); weight tiles arrive via TDM, x via VALU convert.
// ---------------------------------------------------------------------------
__global__ __launch_bounds__(256) void qkv_kernel(
    const float* __restrict__ x,
    const __bf16* __restrict__ WqT, const __bf16* __restrict__ WkT,
    const __bf16* __restrict__ WvT,
    const float* __restrict__ bq, const float* __restrict__ bk,
    const float* __restrict__ bv,
    __bf16* __restrict__ Qt, __bf16* __restrict__ Kt, __bf16* __restrict__ V) {
  __shared__ __bf16 sA[128 * LDST];
  __shared__ __bf16 sB[3][64 * LDST];

  int m0  = blockIdx.x * 128;
  int n0  = blockIdx.y * 64;
  int tid = threadIdx.x, wid = tid >> 5, lane = tid & 31;
  const float* Abase = x + (size_t)m0 * HDIM;
  unsigned int b0off = lds_off_of(sB[0]);
  unsigned int b1off = lds_off_of(sB[1]);
  unsigned int b2off = lds_off_of(sB[2]);

  v8f acc[3][4] = {};
  for (int k0 = 0; k0 < HDIM; k0 += 32) {
    if (wid == 0) {
      tdm_load_2d(b0off, WqT + (size_t)n0 * HDIM + k0, 64, HDIM);
      tdm_load_2d(b1off, WkT + (size_t)n0 * HDIM + k0, 64, HDIM);
      tdm_load_2d(b2off, WvT + (size_t)n0 * HDIM + k0, 64, HDIM);
    }
    tile_f32(sA, Abase + k0, HDIM, 128, tid);
    if (k0 + 32 < HDIM)
      __builtin_prefetch(Abase + (size_t)(tid >> 1) * HDIM + k0 + 32, 0, 1);
    if (wid == 0) __builtin_amdgcn_s_wait_tensorcnt(0);
    __syncthreads();
    v16bf a = load_frag(sA + wid * 16 * LDST, lane);
#pragma unroll
    for (int j = 0; j < 4; ++j) {
      v16bf w0 = load_frag(sB[0] + j * 16 * LDST, lane);
      v16bf w1 = load_frag(sB[1] + j * 16 * LDST, lane);
      v16bf w2 = load_frag(sB[2] + j * 16 * LDST, lane);
      acc[0][j] = wmma_bf16(a, w0, acc[0][j]);
      acc[1][j] = wmma_bf16(a, w1, acc[1][j]);
      acc[2][j] = wmma_bf16(a, w2, acc[2][j]);
    }
    __syncthreads();
  }

  int b      = m0 >> 12;
  int s_base = (m0 & 4095) + wid * 16 + ((lane >> 4) << 3);
  int m_base = m0 + wid * 16 + ((lane >> 4) << 3);
#pragma unroll
  for (int j = 0; j < 4; ++j) {
    int n = n0 + j * 16 + (lane & 15);
    float vbq = bq[n], vbk = bk[n], vbv = bv[n];
#pragma unroll
    for (int i = 0; i < 8; ++i) {
      int s = s_base + i;
      size_t tpos = ((size_t)b * HDIM + n) * SDIM + s;   // (b, d, s)
      Qt[tpos] = (__bf16)(acc[0][j][i] + vbq);
      Kt[tpos] = (__bf16)(acc[1][j][i] + vbk);
      V[(size_t)(m_base + i) * HDIM + n] = (__bf16)(acc[2][j][i] + vbv);
    }
  }
}

// ---------------------------------------------------------------------------
// Shared single-GEMM block body: 128x128 block tile, 8 waves as 4(M) x 2(N),
// each wave computes 32x64 (2 A-frags + 4 B-frags -> 8 WMMAs per K step).
// TDM double-buffered: stage k+1's two tile DMAs are issued before computing
// stage k; s_wait_tensorcnt 2 (+ in-order completion) proves stage k landed.
// ---------------------------------------------------------------------------
template <int KTOT>
__device__ __forceinline__ void gemm_body(const __bf16* Abase, int lda,
                                          const __bf16* Bbase, int ldb,
                                          __bf16 (*sA)[128 * LDST],
                                          __bf16 (*sB)[128 * LDST],
                                          int tid, v8f acc[2][4]) {
  int wid = tid >> 5, lane = tid & 31;
  int wm = wid & 3, wn = wid >> 2;
  unsigned int aoff[2] = {lds_off_of(sA[0]), lds_off_of(sA[1])};
  unsigned int boff[2] = {lds_off_of(sB[0]), lds_off_of(sB[1])};

  if (wid == 0) {                       // prologue: stage 0 in flight
    tdm_load_2d(aoff[0], Abase, 128, lda);
    tdm_load_2d(boff[0], Bbase, 128, ldb);
  }
  for (int k0 = 0; k0 < KTOT; k0 += 32) {
    int cur = (k0 >> 5) & 1;
    if (wid == 0) {
      if (k0 + 32 < KTOT) {             // issue stage k+1 into the other buf
        tdm_load_2d(aoff[cur ^ 1], Abase + k0 + 32, 128, lda);
        tdm_load_2d(boff[cur ^ 1], Bbase + k0 + 32, 128, ldb);
        __builtin_amdgcn_s_wait_tensorcnt(2);   // oldest 2 (stage k) done
      } else {
        __builtin_amdgcn_s_wait_tensorcnt(0);
      }
    }
    __syncthreads();
    const __bf16* cA = sA[cur];
    const __bf16* cB = sB[cur];
    v16bf a0 = load_frag(cA + (wm * 32) * LDST, lane);
    v16bf a1 = load_frag(cA + (wm * 32 + 16) * LDST, lane);
#pragma unroll
    for (int j = 0; j < 4; ++j) {
      v16bf bf = load_frag(cB + (wn * 64 + j * 16) * LDST, lane);
      acc[0][j] = wmma_bf16(a0, bf, acc[0][j]);
      acc[1][j] = wmma_bf16(a1, bf, acc[1][j]);
    }
    __syncthreads();                    // readers done before buf reused
  }
}

// ---------------------------------------------------------------------------
// Kernel 3: scores[b,d,e] = sum_s Qt[b,d,s] * Kt[b,e,s]   (f32 out)
// ---------------------------------------------------------------------------
__global__ __launch_bounds__(256) void scores_kernel(
    const __bf16* __restrict__ Qt, const __bf16* __restrict__ Kt,
    float* __restrict__ scores) {
  __shared__ __bf16 sA[2][128 * LDST];
  __shared__ __bf16 sB[2][128 * LDST];
  int b = blockIdx.z;
  int m0 = blockIdx.x * 128, n0 = blockIdx.y * 128;
  int tid = threadIdx.x, wid = tid >> 5, lane = tid & 31;
  int wm = wid & 3, wn = wid >> 2;

  v8f acc[2][4] = {};
  gemm_body<SDIM>(Qt + ((size_t)b * HDIM + m0) * SDIM, SDIM,
                  Kt + ((size_t)b * HDIM + n0) * SDIM, SDIM, sA, sB, tid, acc);

  int mloc = (lane >> 4) << 3;
#pragma unroll
  for (int mi = 0; mi < 2; ++mi)
#pragma unroll
    for (int j = 0; j < 4; ++j) {
      int e = n0 + wn * 64 + j * 16 + (lane & 15);
#pragma unroll
      for (int i = 0; i < 8; ++i) {
        int d = m0 + wm * 32 + mi * 16 + mloc + i;
        scores[((size_t)b * HDIM + d) * HDIM + e] = acc[mi][j][i];
      }
    }
}

// ---------------------------------------------------------------------------
// Kernel 4: row softmax over e (512 wide), scale = 1/sqrt(512); bf16 out.
// ---------------------------------------------------------------------------
__global__ __launch_bounds__(256) void softmax_kernel(
    const float* __restrict__ scores, __bf16* __restrict__ attn) {
  __shared__ float red[256];
  int row = blockIdx.x, tid = threadIdx.x;
  const float* src = scores + (size_t)row * HDIM;
  const float scale = 0.04419417382415922f;   // 1/sqrt(512)
  float v0 = src[tid] * scale;
  float v1 = src[tid + 256] * scale;
  red[tid] = fmaxf(v0, v1);
  __syncthreads();
  for (int off = 128; off > 0; off >>= 1) {
    if (tid < off) red[tid] = fmaxf(red[tid], red[tid + off]);
    __syncthreads();
  }
  float m = red[0];
  __syncthreads();
  float e0 = __expf(v0 - m), e1 = __expf(v1 - m);
  red[tid] = e0 + e1;
  __syncthreads();
  for (int off = 128; off > 0; off >>= 1) {
    if (tid < off) red[tid] += red[tid + off];
    __syncthreads();
  }
  float inv = 1.0f / red[0];
  attn[(size_t)row * HDIM + tid]       = (__bf16)(e0 * inv);
  attn[(size_t)row * HDIM + tid + 256] = (__bf16)(e1 * inv);
}

// ---------------------------------------------------------------------------
// Kernel 5: ao[b,s,d] = sum_e attn[b,d,e] * V[b,s,e]   (bf16 out, (S,hd))
// ---------------------------------------------------------------------------
__global__ __launch_bounds__(256) void av_kernel(
    const __bf16* __restrict__ attn, const __bf16* __restrict__ V,
    __bf16* __restrict__ ao) {
  __shared__ __bf16 sA[2][128 * LDST];
  __shared__ __bf16 sB[2][128 * LDST];
  int b = blockIdx.z;
  int m0 = blockIdx.x * 128;        // d
  int n0 = blockIdx.y * 128;        // s
  int tid = threadIdx.x, wid = tid >> 5, lane = tid & 31;
  int wm = wid & 3, wn = wid >> 2;

  v8f acc[2][4] = {};
  gemm_body<HDIM>(attn + ((size_t)b * HDIM + m0) * HDIM, HDIM,
                  V + ((size_t)b * SDIM + n0) * HDIM, HDIM, sA, sB, tid, acc);

  int mloc = (lane >> 4) << 3;
#pragma unroll
  for (int mi = 0; mi < 2; ++mi)
#pragma unroll
    for (int j = 0; j < 4; ++j) {
      int s = n0 + wn * 64 + j * 16 + (lane & 15);
#pragma unroll
      for (int i = 0; i < 8; ++i) {
        int d = m0 + wm * 32 + mi * 16 + mloc + i;
        ao[((size_t)b * SDIM + s) * HDIM + d] = (__bf16)acc[mi][j][i];
      }
    }
}

// ---------------------------------------------------------------------------
// Kernel 6: out[m,n] = sum_k ao[m,k] * WoT[n,k] + bo[n]   (f32 out)
// ---------------------------------------------------------------------------
__global__ __launch_bounds__(256) void out_kernel(
    const __bf16* __restrict__ ao, const __bf16* __restrict__ WoT,
    const float* __restrict__ bo, float* __restrict__ out) {
  __shared__ __bf16 sA[2][128 * LDST];
  __shared__ __bf16 sB[2][128 * LDST];
  int m0 = blockIdx.x * 128, n0 = blockIdx.y * 128;
  int tid = threadIdx.x, wid = tid >> 5, lane = tid & 31;
  int wm = wid & 3, wn = wid >> 2;

  v8f acc[2][4] = {};
  gemm_body<HDIM>(ao + (size_t)m0 * HDIM, HDIM,
                  WoT + (size_t)n0 * HDIM, HDIM, sA, sB, tid, acc);

  int mloc = (lane >> 4) << 3;
#pragma unroll
  for (int mi = 0; mi < 2; ++mi)
#pragma unroll
    for (int j = 0; j < 4; ++j) {
      int n = n0 + wn * 64 + j * 16 + (lane & 15);
      float vb = bo[n];
#pragma unroll
      for (int i = 0; i < 8; ++i) {
        int m = m0 + wm * 32 + mi * 16 + mloc + i;
        out[(size_t)m * HDIM + n] = acc[mi][j][i] + vb;
      }
    }
}

// ---------------------------------------------------------------------------
// Launcher.  Workspace layout:
//   [0, 2MB)      4x bf16 transposed weights
//   [2MB, 66MB)   Qt (b,hd,S) bf16   -- reused as ao (b,S,hd) after scores
//   [66MB, 130MB) Kt (b,hd,S) bf16   -- reused as attn bf16 after scores
//   [130MB,194MB) V  (b,S,hd) bf16
//   [194MB,210MB) scores f32 (b,512,512)
// ---------------------------------------------------------------------------
extern "C" void kernel_launch(void* const* d_in, const int* in_sizes, int n_in,
                              void* d_out, int out_size, void* d_ws,
                              size_t ws_size, hipStream_t stream) {
  const float* x  = (const float*)d_in[0];
  const float* Wq = (const float*)d_in[1];
  const float* bq = (const float*)d_in[2];
  const float* Wk = (const float*)d_in[3];
  const float* bk = (const float*)d_in[4];
  const float* Wv = (const float*)d_in[5];
  const float* bv = (const float*)d_in[6];
  const float* Wo = (const float*)d_in[7];
  const float* bo = (const float*)d_in[8];
  float* out = (float*)d_out;

  const size_t WSZ = (size_t)HDIM * HDIM;
  const size_t TSZ = (size_t)NB * HDIM * SDIM;

  __bf16* WqT = (__bf16*)d_ws;
  __bf16* WkT = WqT + WSZ;
  __bf16* WvT = WkT + WSZ;
  __bf16* WoT = WvT + WSZ;
  __bf16* Qt  = WoT + WSZ;
  __bf16* Kt  = Qt + TSZ;
  __bf16* V   = Kt + TSZ;
  float*  scores = (float*)(V + TSZ);
  __bf16* attn = Kt;   // Kt dead after scores_kernel
  __bf16* ao   = Qt;   // Qt dead after scores_kernel

  prep_kernel<<<(HDIM * HDIM) / 256, 256, 0, stream>>>(Wq, Wk, Wv, Wo,
                                                       WqT, WkT, WvT, WoT);
  qkv_kernel<<<dim3(MROWS / 128, HDIM / 64), 256, 0, stream>>>(
      x, WqT, WkT, WvT, bq, bk, bv, Qt, Kt, V);
  scores_kernel<<<dim3(HDIM / 128, HDIM / 128, NB), 256, 0, stream>>>(
      Qt, Kt, scores);
  softmax_kernel<<<NB * HDIM, 256, 0, stream>>>(scores, attn);
  av_kernel<<<dim3(HDIM / 128, SDIM / 128, NB), 256, 0, stream>>>(attn, V, ao);
  out_kernel<<<dim3(MROWS / 128, HDIM / 128), 256, 0, stream>>>(ao, WoT, bo, out);
}
